// IP_FuAd_AttnProcessor2_0_9904194585115
// MI455X (gfx1250) — matrix operand order repaired and checked by reading, not compiled, not run on hardware
//
#include <hip/hip_runtime.h>
#include <hip/hip_bf16.h>

// ---------------- common types / helpers ----------------
typedef __attribute__((ext_vector_type(16))) __bf16 v16bf;
typedef __attribute__((ext_vector_type(16))) short  v16s;
typedef __attribute__((ext_vector_type(8)))  float  v8f;

struct U32x8 { uint4 a, b; };   // 32B carrier for one bf16 fragment

#define S_LEN  4096
#define DMODEL 1280
#define HEADS  20
#define HDIM   64
#define ENC_T  81
#define TXT_T  77

__device__ inline unsigned short f2bf(float f) {
  union { float f; unsigned int u; } c; c.f = f;
  unsigned int u = c.u;
  unsigned int r = u + 0x7FFFu + ((u >> 16) & 1u);  // RNE
  return (unsigned short)(r >> 16);
}

// fragment = two 16-byte runs (p0: elems 0..7, p1: elems 8..15); both 16B aligned
__device__ inline v16bf ld_frag(const unsigned short* p0, const unsigned short* p1) {
  U32x8 u;
  u.a = *(const uint4*)p0;
  u.b = *(const uint4*)p1;
  return __builtin_bit_cast(v16bf, u);
}
// contiguous 32B fragment
__device__ inline v16bf ld_fragc(const unsigned short* p) { return ld_frag(p, p + 8); }

// stage 16 consecutive elements (fp32->bf16 or bf16 passthrough) into LDS (16B aligned)
__device__ inline void stage16(const float* __restrict__ src, unsigned short* dst) {
  const float4* s = (const float4*)src;
  float4 a = s[0], b = s[1], c = s[2], d = s[3];
  v16s t;
  t[0]  = (short)f2bf(a.x); t[1]  = (short)f2bf(a.y);
  t[2]  = (short)f2bf(a.z); t[3]  = (short)f2bf(a.w);
  t[4]  = (short)f2bf(b.x); t[5]  = (short)f2bf(b.y);
  t[6]  = (short)f2bf(b.z); t[7]  = (short)f2bf(b.w);
  t[8]  = (short)f2bf(c.x); t[9]  = (short)f2bf(c.y);
  t[10] = (short)f2bf(c.z); t[11] = (short)f2bf(c.w);
  t[12] = (short)f2bf(d.x); t[13] = (short)f2bf(d.y);
  t[14] = (short)f2bf(d.z); t[15] = (short)f2bf(d.w);
  U32x8 u = __builtin_bit_cast(U32x8, t);
  ((uint4*)dst)[0] = u.a;
  ((uint4*)dst)[1] = u.b;
}
__device__ inline void stage16(const unsigned short* __restrict__ src, unsigned short* dst) {
  const uint4* s = (const uint4*)src;
  ((uint4*)dst)[0] = s[0];
  ((uint4*)dst)[1] = s[1];
}
__device__ inline void stage16zero(unsigned short* dst) {
  uint4 z = {0, 0, 0, 0};
  ((uint4*)dst)[0] = z;
  ((uint4*)dst)[1] = z;
}

// ---------------- generic bf16 WMMA GEMM: C = A(MxK) * B(KxN) ----------------
// block tile 128x128, BK=32, 256 threads (8 waves), each wave 32x64 (2x4 tiles)
// la: A row-major (row, k); lb_t: B K-major (col, k) so fragments are b128 loads
template <typename TA, bool OUT_BF16>
__global__ __launch_bounds__(256) void gemm_wmma(
    const TA* __restrict__ A, const float* __restrict__ B,
    void* __restrict__ C, const float* __restrict__ bias,
    int M, int N, int K)
{
  __shared__ unsigned short la[128][40];    // 80B row stride (16B aligned)
  __shared__ unsigned short lb_t[128][40];  // transposed B tile

  const int tid    = threadIdx.x;
  const int lane   = tid & 31;
  const int wid    = tid >> 5;
  const int lane16 = lane & 15;
  const int kbA    = (lane & 16) ? 16 : 0;  // A-frag K base
  const int kbB    = (lane & 16) ? 16 : 0;  // B-frag K base
  const int bm = blockIdx.y, bn = blockIdx.x;
  const int wm = (wid & 3) * 32;   // 4 waves along M
  const int wn = (wid >> 2) * 64;  // 2 waves along N

  v8f acc[2][4] = {};

  const int ar = tid >> 1;            // 0..127
  const int ac = (tid & 1) * 16;      // 0/16
  const int br = tid >> 3;            // 0..31
  const int bc = (tid & 7) * 16;      // 0..112
  const int arow = bm * 128 + ar;
  const bool aval = arow < M;

  for (int k0 = 0; k0 < K; k0 += 32) {
    // stage A (vectorized, row-major)
    if (aval) stage16(&A[(size_t)arow * K + k0 + ac], &la[ar][ac]);
    else      stage16zero(&la[ar][ac]);
    // stage B (transpose scatter: 16 consecutive cols of row k0+br)
    {
      const float4* s = (const float4*)&B[(size_t)(k0 + br) * N + bn * 128 + bc];
      float4 f0 = s[0], f1 = s[1], f2 = s[2], f3 = s[3];
      float v[16] = {f0.x,f0.y,f0.z,f0.w, f1.x,f1.y,f1.z,f1.w,
                     f2.x,f2.y,f2.z,f2.w, f3.x,f3.y,f3.z,f3.w};
#pragma unroll
      for (int i = 0; i < 16; ++i) lb_t[bc + i][br] = f2bf(v[i]);
    }
    __syncthreads();

    v16bf af[2], bfr[4];
#pragma unroll
    for (int mt = 0; mt < 2; ++mt) {
      const unsigned short* p = &la[wm + mt * 16 + lane16][kbA];
      af[mt] = ld_frag(p, p + 8);
    }
#pragma unroll
    for (int nt = 0; nt < 4; ++nt)
      bfr[nt] = ld_fragc(&lb_t[wn + nt * 16 + lane16][kbB]);
#pragma unroll
    for (int mt = 0; mt < 2; ++mt)
#pragma unroll
      for (int nt = 0; nt < 4; ++nt)
        acc[mt][nt] = __builtin_amdgcn_wmma_f32_16x16x32_bf16(
            false, af[mt], false, bfr[nt], (short)0, acc[mt][nt], false, false);
    __syncthreads();
  }

  const int hi = (lane >> 4) & 1;
#pragma unroll
  for (int mt = 0; mt < 2; ++mt) {
#pragma unroll
    for (int nt = 0; nt < 4; ++nt) {
      const int col = bn * 128 + wn + nt * 16 + lane16;
#pragma unroll
      for (int v = 0; v < 8; ++v) {
        const int row = bm * 128 + wm + mt * 16 + hi * 8 + v;
        if (row < M) {
          if constexpr (OUT_BF16)
            ((unsigned short*)C)[(size_t)row * N + col] = f2bf(acc[mt][nt][v]);
          else
            ((float*)C)[(size_t)row * N + col] =
                acc[mt][nt][v] + (bias ? bias[col] : 0.0f);
        }
      }
    }
  }
}

// ---------------- masked softmax over NT 16-col score tiles ----------------
template <int NT>
__device__ inline void softmax_tiles(v8f* st, int lane16, int validCols) {
  const float SC = 0.125f;  // 1/sqrt(64)
#pragma unroll
  for (int j = 0; j < NT; ++j) {
    const float maskv = ((j * 16 + lane16) < validCols) ? 0.0f : -1e30f;
#pragma unroll
    for (int v = 0; v < 8; ++v) st[j][v] = st[j][v] * SC + maskv;
  }
#pragma unroll
  for (int v = 0; v < 8; ++v) {
    float m = -1e30f;
#pragma unroll
    for (int j = 0; j < NT; ++j) m = fmaxf(m, st[j][v]);
#pragma unroll
    for (int off = 1; off < 16; off <<= 1)
      m = fmaxf(m, __shfl_xor(m, off, 32));
    float sum = 0.0f;
#pragma unroll
    for (int j = 0; j < NT; ++j) { float e = __expf(st[j][v] - m); st[j][v] = e; sum += e; }
#pragma unroll
    for (int off = 1; off < 16; off <<= 1)
      sum += __shfl_xor(sum, off, 32);
    const float inv = 1.0f / sum;
#pragma unroll
    for (int j = 0; j < NT; ++j) st[j][v] *= inv;
  }
}

// ---- attention helpers (all register indices compile-time constant) ----
__device__ inline void load_qfrags(const unsigned short* __restrict__ Q, int bb, int m0,
                                   int h, int lane, int lane16, v16bf& f0, v16bf& f1) {
  const unsigned short* p =
      Q + (size_t)(bb * S_LEN + m0 + lane16) * DMODEL + h * HDIM + ((lane & 16) ? 16 : 0);
  f0 = ld_frag(p, p + 8);         // dims [kb..kb+15]
  f1 = ld_frag(p + 32, p + 40);   // same pattern, second 32-dim block
}

// scores for 5 key tiles (80 keys) from one LDS K bank (key-major rows of 64 dims)
__device__ inline void text_scores(const unsigned short (*ktp)[64], v16bf q0, v16bf q1,
                                   int lane16, int hb, v8f st[5]) {
#pragma unroll
  for (int j = 0; j < 5; ++j) {
    const unsigned short* p = &ktp[j * 16 + lane16][hb];
    v8f a = {};
    a = __builtin_amdgcn_wmma_f32_16x16x32_bf16(false, q0, false, ld_fragc(p),
                                                (short)0, a, false, false);
    a = __builtin_amdgcn_wmma_f32_16x16x32_bf16(false, q1, false, ld_fragc(p + 32),
                                                (short)0, a, false, false);
    st[j] = a;
  }
}

__device__ inline v8f ip_scores(const unsigned short (*kp)[64], v16bf q0, v16bf q1,
                                int lane16, int hb) {
  const unsigned short* p = &kp[lane16][hb];
  v8f a = {};
  a = __builtin_amdgcn_wmma_f32_16x16x32_bf16(false, q0, false, ld_fragc(p),
                                              (short)0, a, false, false);
  a = __builtin_amdgcn_wmma_f32_16x16x32_bf16(false, q1, false, ld_fragc(p + 32),
                                              (short)0, a, false, false);
  return a;
}

// ---------------- fused attention (text + blended IP) ----------------
// grid: (S/64, HEADS, B), 128 threads (4 waves), each wave = 16 query rows
__global__ __launch_bounds__(128) void attn_wmma(
    const unsigned short* __restrict__ Q,  const unsigned short* __restrict__ Kt,
    const unsigned short* __restrict__ Vt, const unsigned short* __restrict__ Kip,
    const unsigned short* __restrict__ Vip, unsigned short* __restrict__ O)
{
  __shared__ unsigned short kt[2][80][64];     // key-major ([0]=own, [1]=pair)
  __shared__ unsigned short vt_t[64][104];     // dim-major: vt_t[d][key], keys padded to 96
  __shared__ unsigned short kip[2][16][64];    // key-major
  __shared__ unsigned short vip_t[64][40];     // dim-major, keys padded to 32
  __shared__ unsigned short pbuf[4][16][96];   // per-wave P re-layout buffer

  const int tid = threadIdx.x;
  const int lane = tid & 31, wid = tid >> 5;
  const int lane16 = lane & 15, hi = (lane >> 4) & 1;
  const int hb  = hi ? 16 : 0;   // B-frag K base
  const int kbA = hi ? 16 : 0;   // A-frag K base
  const int h = blockIdx.y, b = blockIdx.z;
  const bool odd = (b & 1) != 0;

  for (int i = tid; i < 2 * 80 * 64; i += 128) ((unsigned short*)kt)[i] = 0;
  for (int i = tid; i < 64 * 104;    i += 128) ((unsigned short*)vt_t)[i] = 0;
  for (int i = tid; i < 2 * 16 * 64; i += 128) ((unsigned short*)kip)[i] = 0;
  for (int i = tid; i < 64 * 40;     i += 128) ((unsigned short*)vip_t)[i] = 0;
  __syncthreads();

  // own-batch K/V (always)
  for (int i = tid; i < TXT_T * 64; i += 128) {
    int t = i >> 6, d = i & 63;
    kt[0][t][d]  = Kt[((size_t)(b * ENC_T + t)) * DMODEL + h * HDIM + d];
    vt_t[d][t]   = Vt[((size_t)(b * ENC_T + t)) * DMODEL + h * HDIM + d];
  }
  if (odd) {
    for (int i = tid; i < TXT_T * 64; i += 128) {
      int t = i >> 6, d = i & 63;
      kt[1][t][d] = Kt[((size_t)((b - 1) * ENC_T + t)) * DMODEL + h * HDIM + d];
    }
    for (int i = tid; i < 4 * 64; i += 128) {
      int j = i >> 6, d = i & 63;
      kip[0][j][d] = Kip[((size_t)(b * ENC_T + TXT_T + j)) * DMODEL + h * HDIM + d];
      kip[1][j][d] = Kip[((size_t)((b - 1) * ENC_T + TXT_T + j)) * DMODEL + h * HDIM + d];
      vip_t[d][j]  = Vip[((size_t)(b * ENC_T + TXT_T + j)) * DMODEL + h * HDIM + d];
    }
  }
  __syncthreads();

  const int m0 = blockIdx.x * 64 + wid * 16;

  // ---- pair side first (only odd batches): scores -> softmax -> stp ----
  v16bf qp0, qp1;          // pair Q fragments (kept for the IP section)
  v8f stp[5];
  if (odd) {
    load_qfrags(Q, b - 1, m0, h, lane, lane16, qp0, qp1);
    text_scores(kt[1], qp0, qp1, lane16, hb, stp);
    softmax_tiles<5>(stp, lane16, TXT_T);
  }

  // ---- own side: scores -> softmax -> blend (releases stp) ----
  v16bf q0, q1;
  load_qfrags(Q, b, m0, h, lane, lane16, q0, q1);
  v8f st[5];
  text_scores(kt[0], q0, q1, lane16, hb, st);
  softmax_tiles<5>(st, lane16, TXT_T);
  if (odd) {
#pragma unroll
    for (int j = 0; j < 5; ++j)
#pragma unroll
      for (int v = 0; v < 8; ++v)
        st[j][v] = 0.5f * (st[j][v] + stp[j][v]);
  }

  // ---- P re-layout through LDS (releases st), then P@V ----
#pragma unroll
  for (int j = 0; j < 5; ++j)
#pragma unroll
    for (int v = 0; v < 8; ++v)
      pbuf[wid][hi * 8 + v][j * 16 + lane16] = f2bf(st[j][v]);
#pragma unroll
  for (int v = 0; v < 8; ++v) pbuf[wid][hi * 8 + v][80 + lane16] = 0;

  v8f out[4] = {};
#pragma unroll
  for (int ks = 0; ks < 3; ++ks) {
    const unsigned short* pp = &pbuf[wid][lane16][ks * 32 + kbA];
    v16bf pa = ld_frag(pp, pp + 8);
#pragma unroll
    for (int nt = 0; nt < 4; ++nt) {
      v16bf vb = ld_fragc(&vt_t[nt * 16 + lane16][ks * 32 + hb]);
      out[nt] = __builtin_amdgcn_wmma_f32_16x16x32_bf16(false, pa, false, vb,
                                                        (short)0, out[nt], false, false);
    }
  }

  // ---- blended IP attention, only for odd batches ----
  if (odd) {
    v8f sipp = ip_scores(kip[1], qp0, qp1, lane16, hb);
    softmax_tiles<1>(&sipp, lane16, 4);
    v8f sip = ip_scores(kip[0], q0, q1, lane16, hb);
    softmax_tiles<1>(&sip, lane16, 4);
#pragma unroll
    for (int v = 0; v < 8; ++v) sip[v] = 0.5f * (sip[v] + sipp[v]);

#pragma unroll
    for (int v = 0; v < 8; ++v) {
      pbuf[wid][hi * 8 + v][lane16]      = f2bf(sip[v]);
      pbuf[wid][hi * 8 + v][16 + lane16] = 0;
    }
    const unsigned short* pp = &pbuf[wid][lane16][kbA];
    v16bf pa = ld_frag(pp, pp + 8);
#pragma unroll
    for (int nt = 0; nt < 4; ++nt) {
      v16bf vb = ld_fragc(&vip_t[nt * 16 + lane16][hb]);
      out[nt] = __builtin_amdgcn_wmma_f32_16x16x32_bf16(false, pa, false, vb,
                                                        (short)0, out[nt], false, false);
    }
  }

#pragma unroll
  for (int nt = 0; nt < 4; ++nt)
#pragma unroll
    for (int v = 0; v < 8; ++v)
      O[((size_t)(b * S_LEN + m0 + hi * 8 + v)) * DMODEL + h * HDIM + nt * 16 + lane16] =
          f2bf(out[nt][v]);
}

// ---------------- host launcher ----------------
extern "C" void kernel_launch(void* const* d_in, const int* in_sizes, int n_in,
                              void* d_out, int out_size, void* d_ws, size_t ws_size,
                              hipStream_t stream) {
  const float* hidden = (const float*)d_in[0];   // (4,4096,1280)
  const float* enc    = (const float*)d_in[1];   // (4,81,768)
  const float* Wq     = (const float*)d_in[2];   // (1280,1280)
  const float* Wk     = (const float*)d_in[3];   // (768,1280)
  const float* Wv     = (const float*)d_in[4];
  const float* Wkip   = (const float*)d_in[5];
  const float* Wvip   = (const float*)d_in[6];
  const float* Wo     = (const float*)d_in[7];   // (1280,1280)
  const float* bo     = (const float*)d_in[8];   // (1280,)
  float* out = (float*)d_out;

  const size_t QN  = (size_t)4 * S_LEN * DMODEL;
  const size_t KVN = (size_t)4 * ENC_T * DMODEL;

  char* ws = (char*)d_ws;
  unsigned short* Qbuf  = (unsigned short*)ws;              ws += QN * 2;
  unsigned short* Abuf  = (unsigned short*)ws;              ws += QN * 2;
  unsigned short* Kbuf  = (unsigned short*)ws;              ws += KVN * 2;
  unsigned short* Vbuf  = (unsigned short*)ws;              ws += KVN * 2;
  unsigned short* Kibuf = (unsigned short*)ws;              ws += KVN * 2;
  unsigned short* Vibuf = (unsigned short*)ws;              ws += KVN * 2;

  const int Mbig = 4 * S_LEN;       // 16384
  const int Menc = 4 * ENC_T;       // 324

  // K/V/K_ip/V_ip projections (style rows through Wk/Wv computed but unused)
  dim3 gEnc(DMODEL / 128, (Menc + 127) / 128);
  gemm_wmma<float, true><<<gEnc, 256, 0, stream>>>(enc, Wk,   (void*)Kbuf,  nullptr, Menc, DMODEL, 768);
  gemm_wmma<float, true><<<gEnc, 256, 0, stream>>>(enc, Wv,   (void*)Vbuf,  nullptr, Menc, DMODEL, 768);
  gemm_wmma<float, true><<<gEnc, 256, 0, stream>>>(enc, Wkip, (void*)Kibuf, nullptr, Menc, DMODEL, 768);
  gemm_wmma<float, true><<<gEnc, 256, 0, stream>>>(enc, Wvip, (void*)Vibuf, nullptr, Menc, DMODEL, 768);

  // Q projection
  dim3 gBig(DMODEL / 128, Mbig / 128);
  gemm_wmma<float, true><<<gBig, 256, 0, stream>>>(hidden, Wq, (void*)Qbuf, nullptr, Mbig, DMODEL, DMODEL);

  // fused attention
  dim3 gAtt(S_LEN / 64, HEADS, 4);
  attn_wmma<<<gAtt, 128, 0, stream>>>(Qbuf, Kbuf, Vbuf, Kibuf, Vibuf, Abuf);

  // output projection + bias (fp32 out)
  gemm_wmma<unsigned short, false><<<gBig, 256, 0, stream>>>(Abuf, Wo, (void*)out, bo, Mbig, DMODEL, DMODEL);
}
